// DiffusionGraphConv_22127671509774
// MI455X (gfx1250) — matrix operand order repaired and checked by reading, not compiled, not run on hardware
//
#include <hip/hip_runtime.h>
#include <stdint.h>

// ---------------------------------------------------------------------------
// Problem constants (from the reference)
// ---------------------------------------------------------------------------
#define BB   32          // batch
#define NN   4096        // nodes
#define FF   128         // feature width D+H
#define OO   128         // output size
#define MM   5           // number of diffusion matrices (S*K+1)
#define SS   2           // supports
#define EE   65536       // edges per support
#define ROWW (BB * FF)   // 4096: width of one diffusion-matrix row [b*FF + f]
#define NROW (BB * NN)   // 131072 GEMM rows

typedef __bf16 bf16;
typedef __attribute__((ext_vector_type(8)))  __bf16 v8bf;
typedef __attribute__((ext_vector_type(16))) __bf16 v16bf;
typedef __attribute__((ext_vector_type(8)))  float  v8f;

// ---------------------------------------------------------------------------
// K1: build x0 in layout x0[n*4096 + b*128 + f]  (f contiguous per (n,b))
// ---------------------------------------------------------------------------
__global__ __launch_bounds__(256) void build_x0_kernel(
    const float* __restrict__ inputs,   // [B, N*64]
    const float* __restrict__ state,    // [B, N*64]
    float* __restrict__ x0)
{
    size_t t = (size_t)blockIdx.x * 256 + threadIdx.x;   // 16,777,216 total
    int f = (int)(t & 127);
    int b = (int)((t >> 7) & 31);
    int n = (int)(t >> 12);
    float v;
    if (f < 64) v = inputs[((size_t)b * NN + n) * 64 + f];
    else        v = state [((size_t)b * NN + n) * 64 + (f - 64)];
    x0[t] = v;
}

// ---------------------------------------------------------------------------
// K2: weight permute [F*M, O] fp32 -> Wp[m][o][f] bf16 (B-operand friendly)
// ---------------------------------------------------------------------------
__global__ __launch_bounds__(256) void permute_weight_kernel(
    const float* __restrict__ weight, bf16* __restrict__ wp)
{
    int t = blockIdx.x * 256 + threadIdx.x;   // 5*128*128 = 81920
    int f = t & 127;
    int o = (t >> 7) & 127;
    int m = t >> 14;
    wp[((size_t)m * OO + o) * FF + f] = (bf16)weight[((size_t)f * MM + m) * OO + o];
}

// ---------------------------------------------------------------------------
// CSR construction (once per support; reused by both Chebyshev SpMMs)
// ---------------------------------------------------------------------------
__global__ __launch_bounds__(256) void zero_int_kernel(int* __restrict__ p)
{
    p[blockIdx.x * 256 + threadIdx.x] = 0;
}

__global__ __launch_bounds__(256) void count_edges_kernel(
    const int* __restrict__ rows,       // [S*E] flat
    int* __restrict__ counts)           // [S][NN]
{
    int t = blockIdx.x * 256 + threadIdx.x;   // 0 .. S*E-1
    int s = t >> 16;                          // E = 65536
    atomicAdd(&counts[s * NN + rows[t]], 1);
}

// One block per support: exclusive scan of 4096 counts -> offs[4097] + cursor
__global__ __launch_bounds__(256) void scan_kernel(
    const int* __restrict__ counts,     // [S][NN]
    int* __restrict__ offs,             // [S][NN+1]
    int* __restrict__ cursor)           // [S][NN]
{
    __shared__ int part[256];
    const int s   = blockIdx.x;
    const int tid = threadIdx.x;
    const int base = tid * 16;
    int local[16];
    int sum = 0;
    #pragma unroll
    for (int i = 0; i < 16; ++i) { local[i] = counts[s * NN + base + i]; sum += local[i]; }
    part[tid] = sum;
    __syncthreads();
    for (int off = 1; off < 256; off <<= 1) {
        int add = (tid >= off) ? part[tid - off] : 0;
        __syncthreads();
        part[tid] += add;
        __syncthreads();
    }
    int excl = part[tid] - sum;   // exclusive prefix of this thread's chunk
    #pragma unroll
    for (int i = 0; i < 16; ++i) {
        offs  [s * (NN + 1) + base + i] = excl;
        cursor[s * NN       + base + i] = excl;
        excl += local[i];
    }
    if (tid == 255) offs[s * (NN + 1) + NN] = excl;
}

__global__ __launch_bounds__(256) void scatter_edges_kernel(
    const int* __restrict__ rows,       // [S*E]
    int* __restrict__ cursor,           // [S][NN]
    int* __restrict__ perm)             // [S][E]
{
    int t = blockIdx.x * 256 + threadIdx.x;
    int s = t >> 16;
    int e = t & (EE - 1);
    int pos = atomicAdd(&cursor[s * NN + rows[t]], 1);
    perm[s * EE + pos] = e;
}

// ---------------------------------------------------------------------------
// K3: CSR-gather SpMM with fused Chebyshev update (no fp atomics):
//   y[r,:] = scale * sum_{e in row r} vals[e] * x[cols[e],:]  (- x0[r,:] if cheb)
// One block per row; 256 threads hold the 4096-wide row in 4 float4 regs.
// Gathered x rows (64 MB) are L2-resident on MI455X (192 MB L2).
// ---------------------------------------------------------------------------
__global__ __launch_bounds__(256) void spmm_csr_kernel(
    const int*   __restrict__ offs,     // [NN+1] (this support)
    const int*   __restrict__ perm,     // [E]
    const int*   __restrict__ cols,     // [E]
    const float* __restrict__ vals,     // [E]
    const float* __restrict__ x,
    const float* __restrict__ xprev,    // subtrahend for Chebyshev (may be unused)
    float*       __restrict__ y,
    float scale, int cheb)
{
    const int r   = blockIdx.x;
    const int tid = threadIdx.x;
    const int beg = offs[r];
    const int end = offs[r + 1];

    float4 acc[4];
    #pragma unroll
    for (int k = 0; k < 4; ++k) acc[k] = make_float4(0.f, 0.f, 0.f, 0.f);

    for (int e = beg; e < end; ++e) {
        const int   eid = perm[e];          // uniform across block -> broadcast
        const int   c   = cols[eid];
        const float v   = vals[eid];
        const float* __restrict__ src = x + (size_t)c * ROWW + tid * 4;
        #pragma unroll
        for (int k = 0; k < 4; ++k) {
            float4 sv = *(const float4*)(src + k * 1024);
            acc[k].x += v * sv.x; acc[k].y += v * sv.y;
            acc[k].z += v * sv.z; acc[k].w += v * sv.w;
        }
    }

    float* __restrict__ dst = y + (size_t)r * ROWW + tid * 4;
    const float* __restrict__ prv = xprev + (size_t)r * ROWW + tid * 4;
    #pragma unroll
    for (int k = 0; k < 4; ++k) {
        float4 o;
        o.x = scale * acc[k].x; o.y = scale * acc[k].y;
        o.z = scale * acc[k].z; o.w = scale * acc[k].w;
        if (cheb) {
            float4 p = *(const float4*)(prv + k * 1024);
            o.x -= p.x; o.y -= p.y; o.z -= p.z; o.w -= p.w;
        }
        *(float4*)(dst + k * 1024) = o;
    }
}

// ---------------------------------------------------------------------------
// K4: bf16-WMMA GEMM over nm diffusion matrices:
//   out[r, o] (+)= sum_{mi,f} A_mi[n(r)][b(r)*128 + f] * Wp[mi][o][f] (+ bias)
// Block = 128x128 output tile; 8 waves; wave = 16-row strip x 8 WMMA subtiles.
// LDS rows padded to 40 halfs (80B, 16B-aligned) so each lane fragment is two
// ds_load_b128 matching the 16-bit WMMA A/B VGPR layouts.
// ---------------------------------------------------------------------------
__global__ __launch_bounds__(256) void gemm_wmma(
    const float* __restrict__ A0, const float* __restrict__ A1,
    const float* __restrict__ A2, const float* __restrict__ A3,
    const float* __restrict__ A4, int nm,
    const bf16*  __restrict__ wp,    // base of [nm][128 o][128 f] bf16
    const float* __restrict__ bias,  // [128]
    float*       __restrict__ out,   // [131072][128]
    int accumulate)
{
    __shared__ bf16 sA[128 * 40];
    __shared__ bf16 sW[128 * 40];

    const int tid  = threadIdx.x;
    const int lane = tid & 31;
    const int w    = tid >> 5;          // wave 0..7
    const int R0   = blockIdx.x * 128;  // first output row of this tile
    const int bidx = R0 >> 12;          // batch index (constant in tile)
    const int n0   = R0 & 4095;         // first node index
    const int nloc = lane & 15;
    const int hi   = lane >> 4;         // half-wave select

    v8f c[8];
    if (accumulate) {
        #pragma unroll
        for (int ob = 0; ob < 8; ++ob)
            #pragma unroll
            for (int j = 0; j < 8; ++j)
                c[ob][j] = out[(size_t)(R0 + w * 16 + hi * 8 + j) * OO + ob * 16 + nloc];
    } else {
        #pragma unroll
        for (int ob = 0; ob < 8; ++ob) {
            float bv = bias[ob * 16 + nloc];
            #pragma unroll
            for (int j = 0; j < 8; ++j) c[ob][j] = bv;
        }
    }

    for (int mi = 0; mi < nm; ++mi) {
        const float* __restrict__ A =
            (mi == 0) ? A0 : (mi == 1) ? A1 : (mi == 2) ? A2 : (mi == 3) ? A3 : A4;
        const bf16* __restrict__ Wm = wp + (size_t)mi * (OO * FF);

        #pragma unroll
        for (int fc = 0; fc < 4; ++fc) {          // K chunks of 32 (F = 128)
            __syncthreads();
            // --- stage A tile: 128 rows x 32 f, fp32 -> bf16 ---
            {
                int rbase = tid >> 3;             // 0..31
                int col   = (tid & 7) * 4;        // 0..28
                #pragma unroll
                for (int p = 0; p < 4; ++p) {
                    int row = p * 32 + rbase;
                    float4 v = *(const float4*)(A + (size_t)(n0 + row) * ROWW
                                                  + bidx * FF + fc * 32 + col);
                    bf16* d = &sA[row * 40 + col];
                    d[0] = (bf16)v.x; d[1] = (bf16)v.y; d[2] = (bf16)v.z; d[3] = (bf16)v.w;
                }
            }
            // --- stage W tile: sW[o][fi] = Wm[o][fc*32 + fi] ---
            {
                int o  = tid >> 1;
                int fh = (tid & 1) * 16;
                const v8bf* src = (const v8bf*)(Wm + (size_t)o * FF + fc * 32 + fh);
                v8bf w0 = src[0];
                v8bf w1 = src[1];
                *(v8bf*)(&sW[o * 40 + fh])     = w0;
                *(v8bf*)(&sW[o * 40 + fh + 8]) = w1;
            }
            __syncthreads();

            // A fragment (16x32): lanes 0-15 -> K 0-7|16-23, lanes 16-31 -> K 8-15|24-31
            const int arow = w * 16 + nloc;
            v8bf alo = *(const v8bf*)(&sA[arow * 40 + hi * 8]);
            v8bf ahi = *(const v8bf*)(&sA[arow * 40 + 16 + hi * 8]);
            v16bf a = __builtin_shufflevector(alo, ahi,
                        0,1,2,3,4,5,6,7,8,9,10,11,12,13,14,15);

            #pragma unroll
            for (int ob = 0; ob < 8; ++ob) {
                // B fragment (32x16): lanes 0-15 -> K 0-7|16-23, lanes 16-31 -> K 8-15|24-31
                const int orow = ob * 16 + nloc;
                v8bf blo = *(const v8bf*)(&sW[orow * 40 + hi * 8]);
                v8bf bhi = *(const v8bf*)(&sW[orow * 40 + 16 + hi * 8]);
                v16bf bfr = __builtin_shufflevector(blo, bhi,
                            0,1,2,3,4,5,6,7,8,9,10,11,12,13,14,15);
                c[ob] = __builtin_amdgcn_wmma_f32_16x16x32_bf16(
                            false, a, false, bfr, (short)0, c[ob], false, false);
            }
        }
    }

    #pragma unroll
    for (int ob = 0; ob < 8; ++ob)
        #pragma unroll
        for (int j = 0; j < 8; ++j)
            out[(size_t)(R0 + w * 16 + hi * 8 + j) * OO + ob * 16 + nloc] = c[ob][j];
}

// ---------------------------------------------------------------------------
// Host-side orchestration (graph-capture safe: kernels only, all on stream)
// ---------------------------------------------------------------------------
extern "C" void kernel_launch(void* const* d_in, const int* in_sizes, int n_in,
                              void* d_out, int out_size, void* d_ws, size_t ws_size,
                              hipStream_t stream)
{
    const float* inputs = (const float*)d_in[0];
    const float* state  = (const float*)d_in[1];
    const int*   rows   = (const int*)  d_in[2];   // [S, E]
    const int*   cols   = (const int*)  d_in[3];
    const float* vals   = (const float*)d_in[4];
    const float* weight = (const float*)d_in[5];   // [F*M, O]
    const float* bias   = (const float*)d_in[6];   // [O]
    float* out = (float*)d_out;

    const size_t MAT  = (size_t)NN * ROWW;          // 16,777,216 floats / matrix
    const size_t WPH  = (size_t)MM * OO * FF;       // 81,920 bf16
    const size_t AUXI = (size_t)SS * NN             // counts
                      + (size_t)SS * (NN + 1)       // offs
                      + (size_t)SS * NN             // cursor
                      + (size_t)SS * EE;            // perm
    const size_t need5 = 5 * MAT * 4 + WPH * 2 + AUXI * 4 + 64;
    const int big = (ws_size >= need5) ? 1 : 0;
    const int nbuf = big ? 5 : 3;

    float* bufs = (float*)d_ws;                      // nbuf matrices
    bf16*  wp   = (bf16*)(bufs + (size_t)nbuf * MAT);
    int*   counts = (int*)(wp + WPH);
    int*   offs   = counts + (size_t)SS * NN;
    int*   cursor = offs   + (size_t)SS * (NN + 1);
    int*   perm   = cursor + (size_t)SS * NN;

    float* x0 = bufs;
    const int ELT_BLOCKS = (int)(MAT / 256);         // 65536

    // 1) inputs: x0, permuted weight, CSR for both supports
    build_x0_kernel<<<ELT_BLOCKS, 256, 0, stream>>>(inputs, state, x0);
    permute_weight_kernel<<<(int)(WPH / 256), 256, 0, stream>>>(weight, wp);
    zero_int_kernel<<<(SS * NN) / 256, 256, 0, stream>>>(counts);
    count_edges_kernel<<<(SS * EE) / 256, 256, 0, stream>>>(rows, counts);
    scan_kernel<<<SS, 256, 0, stream>>>(counts, offs, cursor);
    scatter_edges_kernel<<<(SS * EE) / 256, 256, 0, stream>>>(rows, cursor, perm);

    if (big) {
        // 2) all 4 SpMM gathers, then one fused 5-matrix WMMA GEMM
        float* x1a = bufs + 1 * MAT;  float* x2a = bufs + 2 * MAT;
        float* x1b = bufs + 3 * MAT;  float* x2b = bufs + 4 * MAT;
        for (int s = 0; s < SS; ++s) {
            const int* of = offs + (size_t)s * (NN + 1);
            const int* pm = perm + (size_t)s * EE;
            const int* cs = cols + (size_t)s * EE;
            const float* vs = vals + (size_t)s * EE;
            float* x1 = s ? x1b : x1a;
            float* x2 = s ? x2b : x2a;
            spmm_csr_kernel<<<NN, 256, 0, stream>>>(of, pm, cs, vs, x0, x0, x1, 1.f, 0);
            spmm_csr_kernel<<<NN, 256, 0, stream>>>(of, pm, cs, vs, x1, x0, x2, 2.f, 1);
        }
        gemm_wmma<<<NROW / 128, 256, 0, stream>>>(x0, x1a, x2a, x1b, x2b, 5,
                                                  wp, bias, out, 0);
    } else {
        // 2') 3-buffer fallback: accumulate GEMM pass per diffusion matrix
        float* x1 = bufs + 1 * MAT;
        float* x2 = bufs + 2 * MAT;
        gemm_wmma<<<NROW / 128, 256, 0, stream>>>(x0, x0, x0, x0, x0, 1,
                                                  wp, bias, out, 0);
        for (int s = 0; s < SS; ++s) {
            const int* of = offs + (size_t)s * (NN + 1);
            const int* pm = perm + (size_t)s * EE;
            const int* cs = cols + (size_t)s * EE;
            const float* vs = vals + (size_t)s * EE;
            const int m1 = 1 + 2 * s, m2 = 2 + 2 * s;
            spmm_csr_kernel<<<NN, 256, 0, stream>>>(of, pm, cs, vs, x0, x0, x1, 1.f, 0);
            gemm_wmma<<<NROW / 128, 256, 0, stream>>>(x1, x1, x1, x1, x1, 1,
                                                      wp + (size_t)m1 * OO * FF, bias, out, 1);
            spmm_csr_kernel<<<NN, 256, 0, stream>>>(of, pm, cs, vs, x1, x0, x2, 2.f, 1);
            gemm_wmma<<<NROW / 128, 256, 0, stream>>>(x2, x2, x2, x2, x2, 1,
                                                      wp + (size_t)m2 * OO * FF, bias, out, 1);
        }
    }
}
